// GINConvWithGlobal_1597727834590
// MI455X (gfx1250) — matrix operand order repaired and checked by reading, not compiled, not run on hardware
//
#include <hip/hip_runtime.h>
#include <hip/hip_bf16.h>
#include <math.h>

#define NNODES 50000
#define NEDGES 800000
#define DIM 128
#define BN_EPS 1e-5f

typedef __attribute__((ext_vector_type(16))) _Float16 v16h;
typedef __attribute__((ext_vector_type(8)))  float    v8f;

union Frag16 { v16h v; uint4 q[2]; };
union H4     { _Float16 h[4]; uint2 u; };

// ---------------------------------------------------------------------------
// Weight prep: fp32 row-major W[k][n] -> f16 fragments in WMMA B layout.
// Fragment layout: frag[((kc*8 + ntile)*32 + lane)*16 + i]
//   holds W[kc*32 + 16*(lane>>4) + i][ntile*16 + (lane&15)]
// so each lane's v16h B fragment is 32 contiguous bytes.
// ---------------------------------------------------------------------------
__global__ __launch_bounds__(256) void wprep_kernel(
    const float* __restrict__ Wa1, const float* __restrict__ Wa2,
    const float* __restrict__ Wb1, const float* __restrict__ Wb2,
    _Float16* __restrict__ wf)
{
  const float* W = (blockIdx.x == 0) ? Wa1 : (blockIdx.x == 1) ? Wa2
                 : (blockIdx.x == 2) ? Wb1 : Wb2;
  _Float16* out = wf + (size_t)blockIdx.x * 16384;
  for (int idx = threadIdx.x; idx < 1024; idx += 256) {
    const int lane = idx & 31;
    const int t    = (idx >> 5) & 7;
    const int kc   = idx >> 8;
    const int n    = t * 16 + (lane & 15);
    const int kb   = kc * 32 + 16 * (lane >> 4);
    _Float16* o = out + (size_t)idx * 16;
    #pragma unroll
    for (int i = 0; i < 16; ++i)
      o[i] = (_Float16)W[(size_t)(kb + i) * DIM + n];
  }
}

// ---------------------------------------------------------------------------
// One GEMM layer: 16x128 f16 A tile (LDS) x 128x128 f16 W -> 8 v8f tiles.
// B fragments for each k-chunk are batched into registers so the loads issue
// as one clause and overlap with the WMMAs instead of serializing.
// ---------------------------------------------------------------------------
__device__ __forceinline__ void wave_gemm_layer(
    const _Float16* A, const _Float16* __restrict__ wf,
    int lane, int g, int ln, v8f acc[8])
{
  #pragma unroll
  for (int kc = 0; kc < 4; ++kc) {
    Frag16 a;
    a.q[0] = *(const uint4*)&A[ln * DIM + kc * 32 + 16 * g];
    a.q[1] = *(const uint4*)&A[ln * DIM + kc * 32 + 16 * g + 8];
    Frag16 b[8];
    #pragma unroll
    for (int t = 0; t < 8; ++t) {
      const uint4* bq =
          (const uint4*)(wf + ((size_t)((kc * 8 + t) * 32 + lane)) * 16);
      b[t].q[0] = bq[0];
      b[t].q[1] = bq[1];
    }
    #pragma unroll
    for (int t = 0; t < 8; ++t)
      acc[t] = __builtin_amdgcn_wmma_f32_16x16x32_f16(
          false, a.v, false, b[t].v, (short)0, acc[t], false, false);
  }
}

// ---------------------------------------------------------------------------
// Per-wave 2-layer MLP on a 16x128 f16 tile staged in LDS (A).
// ---------------------------------------------------------------------------
__device__ __forceinline__ void wave_mlp2(
    _Float16* A,
    const _Float16* __restrict__ w1f, const _Float16* __restrict__ w2f,
    const float* __restrict__ b1, const float* __restrict__ b2,
    int lane, v8f out[8])
{
  const int g  = lane >> 4;
  const int ln = lane & 15;

  v8f acc[8];
  #pragma unroll
  for (int t = 0; t < 8; ++t) {
    const float bv = b1[t * 16 + ln];
    #pragma unroll
    for (int i = 0; i < 8; ++i) acc[t][i] = bv;
  }
  wave_gemm_layer(A, w1f, lane, g, ln, acc);

  __syncthreads();
  // ReLU, restage as f16 A-tile for layer 2
  #pragma unroll
  for (int t = 0; t < 8; ++t) {
    const int n = t * 16 + ln;
    #pragma unroll
    for (int r = 0; r < 8; ++r)
      A[(r + 8 * g) * DIM + n] = (_Float16)fmaxf(acc[t][r], 0.0f);
  }
  __syncthreads();

  #pragma unroll
  for (int t = 0; t < 8; ++t) {
    const float bv = b2[t * 16 + ln];
    #pragma unroll
    for (int i = 0; i < 8; ++i) out[t][i] = bv;
  }
  wave_gemm_layer(A, w2f, lane, g, ln, out);
}

// ---------------------------------------------------------------------------
// Edge MLP: me = h[src]+h[dst]+e, y = mlp_b(me)  (raw, pre-BN) -> out_e slot.
// Also emits per-block column sum / sumsq partials (deterministic).
// ---------------------------------------------------------------------------
__global__ __launch_bounds__(128) void edge_mlp_kernel(
    const float* __restrict__ h, const float* __restrict__ e,
    const int* __restrict__ src, const int* __restrict__ dst,
    const _Float16* __restrict__ w1f, const _Float16* __restrict__ w2f,
    const float* __restrict__ b1, const float* __restrict__ b2,
    float* __restrict__ yout, float* __restrict__ partials)
{
  __shared__ _Float16 ldsA[4][16 * DIM];
  __shared__ float wsum[4][DIM];
  __shared__ float wsq[4][DIM];

  const int tid  = threadIdx.x;
  const int wave = tid >> 5;
  const int lane = tid & 31;
  const int g = lane >> 4, ln = lane & 15;
  _Float16* A = ldsA[wave];
  const long rowBase = ((long)blockIdx.x * 4 + wave) * 16;

  { // stage me into LDS as f16 (16 rows x 128 cols, row-major)
    const int r  = lane >> 1;
    const int cb = (lane & 1) * 64;
    const long er = rowBase + r;
    const float* hs = h + (size_t)src[er] * DIM;
    const float* hd = h + (size_t)dst[er] * DIM;
    const float* ep = e + (size_t)er * DIM;
    #pragma unroll
    for (int j = 0; j < 16; ++j) {
      const int c = cb + 4 * j;
      float4 a = *(const float4*)(hs + c);
      float4 b = *(const float4*)(hd + c);
      float4 x = *(const float4*)(ep + c);
      H4 p;
      p.h[0] = (_Float16)(a.x + b.x + x.x);
      p.h[1] = (_Float16)(a.y + b.y + x.y);
      p.h[2] = (_Float16)(a.z + b.z + x.z);
      p.h[3] = (_Float16)(a.w + b.w + x.w);
      *(uint2*)&A[r * DIM + c] = p.u;
    }
  }
  __syncthreads();

  v8f acc[8];
  wave_mlp2(A, w1f, w2f, b1, b2, lane, acc);

  #pragma unroll
  for (int t = 0; t < 8; ++t) {
    const int n = t * 16 + ln;
    float s = 0.f, ss = 0.f;
    #pragma unroll
    for (int r = 0; r < 8; ++r) {
      const long row = rowBase + r + 8 * g;
      const float v = acc[t][r];
      yout[(size_t)row * DIM + n] = v;
      s += v; ss += v * v;
    }
    s  += __shfl_xor(s, 16);
    ss += __shfl_xor(ss, 16);
    if (g == 0) { wsum[wave][n] = s; wsq[wave][n] = ss; }
  }
  __syncthreads();
  if (tid < DIM) {
    float s  = wsum[0][tid] + wsum[1][tid] + wsum[2][tid] + wsum[3][tid];
    float ss = wsq[0][tid]  + wsq[1][tid]  + wsq[2][tid]  + wsq[3][tid];
    partials[(size_t)blockIdx.x * 256 + tid]       = s;
    partials[(size_t)blockIdx.x * 256 + DIM + tid] = ss;
  }
}

// ---------------------------------------------------------------------------
// Node MLP: x = h + agg, z = mlp_a(x) (raw, pre-BN) -> out_h slot, + partials.
// Tail rows (>= NNODES) are clamped for loads and masked from stores/stats.
// ---------------------------------------------------------------------------
__global__ __launch_bounds__(128) void node_mlp_kernel(
    const float* __restrict__ h, const float* __restrict__ agg,
    const _Float16* __restrict__ w1f, const _Float16* __restrict__ w2f,
    const float* __restrict__ b1, const float* __restrict__ b2,
    float* __restrict__ yout, float* __restrict__ partials)
{
  __shared__ _Float16 ldsA[4][16 * DIM];
  __shared__ float wsum[4][DIM];
  __shared__ float wsq[4][DIM];

  const int tid  = threadIdx.x;
  const int wave = tid >> 5;
  const int lane = tid & 31;
  const int g = lane >> 4, ln = lane & 15;
  _Float16* A = ldsA[wave];
  const long rowBase = ((long)blockIdx.x * 4 + wave) * 16;

  {
    const int r  = lane >> 1;
    const int cb = (lane & 1) * 64;
    long row = rowBase + r;
    if (row >= NNODES) row = NNODES - 1;
    const float* hp = h   + (size_t)row * DIM;
    const float* ap = agg + (size_t)row * DIM;
    #pragma unroll
    for (int j = 0; j < 16; ++j) {
      const int c = cb + 4 * j;
      float4 a = *(const float4*)(hp + c);
      float4 b = *(const float4*)(ap + c);
      H4 p;
      p.h[0] = (_Float16)(a.x + b.x);
      p.h[1] = (_Float16)(a.y + b.y);
      p.h[2] = (_Float16)(a.z + b.z);
      p.h[3] = (_Float16)(a.w + b.w);
      *(uint2*)&A[r * DIM + c] = p.u;
    }
  }
  __syncthreads();

  v8f acc[8];
  wave_mlp2(A, w1f, w2f, b1, b2, lane, acc);

  #pragma unroll
  for (int t = 0; t < 8; ++t) {
    const int n = t * 16 + ln;
    float s = 0.f, ss = 0.f;
    #pragma unroll
    for (int r = 0; r < 8; ++r) {
      const long row = rowBase + r + 8 * g;
      const float v = acc[t][r];
      if (row < NNODES) {
        yout[(size_t)row * DIM + n] = v;
        s += v; ss += v * v;
      }
    }
    s  += __shfl_xor(s, 16);
    ss += __shfl_xor(ss, 16);
    if (g == 0) { wsum[wave][n] = s; wsq[wave][n] = ss; }
  }
  __syncthreads();
  if (tid < DIM) {
    float s  = wsum[0][tid] + wsum[1][tid] + wsum[2][tid] + wsum[3][tid];
    float ss = wsq[0][tid]  + wsq[1][tid]  + wsq[2][tid]  + wsq[3][tid];
    partials[(size_t)blockIdx.x * 256 + tid]       = s;
    partials[(size_t)blockIdx.x * 256 + DIM + tid] = ss;
  }
}

// ---------------------------------------------------------------------------
// Reduce per-block partials: one block per column c (0..255).
// ---------------------------------------------------------------------------
__global__ __launch_bounds__(256) void colreduce_kernel(
    const float* __restrict__ part, int nblk, float* __restrict__ S)
{
  __shared__ float red[256];
  const int c = blockIdx.x;
  float s = 0.f;
  for (int b = threadIdx.x; b < nblk; b += 256)
    s += part[(size_t)b * 256 + c];
  red[threadIdx.x] = s;
  __syncthreads();
  for (int st = 128; st > 0; st >>= 1) {
    if (threadIdx.x < st) red[threadIdx.x] += red[threadIdx.x + st];
    __syncthreads();
  }
  if (threadIdx.x == 0) S[c] = red[0];
}

__global__ void finalize_kernel(const float* __restrict__ S, float inv,
                                float* __restrict__ murs)
{
  const int c = threadIdx.x;  // 0..127
  const float mu  = S[c] * inv;
  const float var = S[c + 128] * inv - mu * mu;
  murs[c]        = mu;
  murs[c + 128]  = rsqrtf(fmaxf(var, 0.0f) + BN_EPS);
}

// ---------------------------------------------------------------------------
// Edge normalize in place: e_new = (y-mu)*rs*gamma + beta + e; scatter into agg.
// ---------------------------------------------------------------------------
__global__ __launch_bounds__(256) void edge_norm_kernel(
    const float* __restrict__ murs, const float* __restrict__ gamma,
    const float* __restrict__ beta, const float* __restrict__ e,
    const int* __restrict__ dst, float* __restrict__ eout,
    float* __restrict__ agg)
{
  const size_t tid = (size_t)blockIdx.x * 256 + threadIdx.x;
  const size_t row = tid >> 5;
  const int c0 = (int)(tid & 31) * 4;
  float4 y = *(const float4*)(eout + row * DIM + c0);
  float4 r = *(const float4*)(e    + row * DIM + c0);
  float4 o;
  o.x = (y.x - murs[c0 + 0]) * murs[DIM + c0 + 0] * gamma[c0 + 0] + beta[c0 + 0] + r.x;
  o.y = (y.y - murs[c0 + 1]) * murs[DIM + c0 + 1] * gamma[c0 + 1] + beta[c0 + 1] + r.y;
  o.z = (y.z - murs[c0 + 2]) * murs[DIM + c0 + 2] * gamma[c0 + 2] + beta[c0 + 2] + r.z;
  o.w = (y.w - murs[c0 + 3]) * murs[DIM + c0 + 3] * gamma[c0 + 3] + beta[c0 + 3] + r.w;
  *(float4*)(eout + row * DIM + c0) = o;
  float* ap = agg + (size_t)dst[row] * DIM + c0;
  atomicAdd(ap + 0, o.x);
  atomicAdd(ap + 1, o.y);
  atomicAdd(ap + 2, o.z);
  atomicAdd(ap + 3, o.w);
}

__global__ __launch_bounds__(256) void node_norm_kernel(
    const float* __restrict__ murs, const float* __restrict__ gamma,
    const float* __restrict__ beta, const float* __restrict__ h,
    float* __restrict__ hout)
{
  const size_t tid = (size_t)blockIdx.x * 256 + threadIdx.x;
  const size_t row = tid >> 5;
  const int c0 = (int)(tid & 31) * 4;
  float4 y = *(const float4*)(hout + row * DIM + c0);
  float4 r = *(const float4*)(h    + row * DIM + c0);
  float4 o;
  o.x = (y.x - murs[c0 + 0]) * murs[DIM + c0 + 0] * gamma[c0 + 0] + beta[c0 + 0] + r.x;
  o.y = (y.y - murs[c0 + 1]) * murs[DIM + c0 + 1] * gamma[c0 + 1] + beta[c0 + 1] + r.y;
  o.z = (y.z - murs[c0 + 2]) * murs[DIM + c0 + 2] * gamma[c0 + 2] + beta[c0 + 2] + r.z;
  o.w = (y.w - murs[c0 + 3]) * murs[DIM + c0 + 3] * gamma[c0 + 3] + beta[c0 + 3] + r.w;
  *(float4*)(hout + row * DIM + c0) = o;
}

// ---------------------------------------------------------------------------
extern "C" void kernel_launch(void* const* d_in, const int* in_sizes, int n_in,
                              void* d_out, int out_size, void* d_ws, size_t ws_size,
                              hipStream_t stream)
{
  const float* h   = (const float*)d_in[0];
  const float* e   = (const float*)d_in[1];
  const int*   src = (const int*)d_in[2];
  const int*   dst = (const int*)d_in[3];
  const float* Wa1 = (const float*)d_in[4];
  const float* ba1 = (const float*)d_in[5];
  const float* Wa2 = (const float*)d_in[6];
  const float* ba2 = (const float*)d_in[7];
  const float* Wb1 = (const float*)d_in[8];
  const float* bb1 = (const float*)d_in[9];
  const float* Wb2 = (const float*)d_in[10];
  const float* bb2 = (const float*)d_in[11];
  const float* ga  = (const float*)d_in[12];
  const float* bea = (const float*)d_in[13];
  const float* gb  = (const float*)d_in[14];
  const float* beb = (const float*)d_in[15];
  (void)in_sizes; (void)n_in; (void)out_size; (void)ws_size;

  float* out_h = (float*)d_out;                       // [NNODES, DIM] h_new
  float* out_e = out_h + (size_t)NNODES * DIM;        // [NEDGES, DIM] e_new

  // workspace layout
  char* ws = (char*)d_ws;
  _Float16* wfrag = (_Float16*)ws;                     // 4 * 16384 halves (128 KB)
  float* S_e    = (float*)(ws + 131072);               // 256 f
  float* S_h    = S_e + 256;                           // 256 f
  float* murs_e = S_h + 256;                           // 256 f (mu | rsig)
  float* murs_h = murs_e + 256;                        // 256 f
  float* agg    = (float*)(ws + 135168);               // NNODES*DIM f (25.6 MB)
  float* part_e = (float*)(ws + 25735168ull);          // 12500*256 f (12.8 MB)
  float* part_h = (float*)(ws + 38535168ull);          // 782*256 f

  const int edgeBlocks = NEDGES / 64;                  // 12500
  const int nodeBlocks = (NNODES + 63) / 64;           // 782

  hipMemsetAsync(agg, 0, (size_t)NNODES * DIM * sizeof(float), stream);
  wprep_kernel<<<4, 256, 0, stream>>>(Wa1, Wa2, Wb1, Wb2, wfrag);

  // --- edge (bond) update ---
  edge_mlp_kernel<<<edgeBlocks, 128, 0, stream>>>(
      h, e, src, dst, wfrag + 2 * 16384, wfrag + 3 * 16384, bb1, bb2,
      out_e, part_e);
  colreduce_kernel<<<256, 256, 0, stream>>>(part_e, edgeBlocks, S_e);
  finalize_kernel<<<1, 128, 0, stream>>>(S_e, 1.0f / (float)NEDGES, murs_e);
  edge_norm_kernel<<<(NEDGES * 32) / 256, 256, 0, stream>>>(
      murs_e, gb, beb, e, dst, out_e, agg);

  // --- node (atom) update ---
  node_mlp_kernel<<<nodeBlocks, 128, 0, stream>>>(
      h, agg, wfrag, wfrag + 16384, ba1, ba2, out_h, part_h);
  colreduce_kernel<<<256, 256, 0, stream>>>(part_h, nodeBlocks, S_h);
  finalize_kernel<<<1, 128, 0, stream>>>(S_h, 1.0f / (float)NNODES, murs_h);
  node_norm_kernel<<<(NNODES * 32) / 256, 256, 0, stream>>>(
      murs_h, ga, bea, h, out_h);
}
